// NonParametricCritic_16338055594570
// MI455X (gfx1250) — compile-verified, matrix operand order checked by reading, not run on hardware
//
#include <hip/hip_runtime.h>
#include <math.h>

// ---------------- problem constants ----------------
#define OBS_DIM 64
#define ACT_DIM 12
#define IN_DIM  76
#define HID     256
#define BATCH   1024
#define CAP     100000
#define MT      (BATCH / 16)          // 64 batch tiles of 16 rows
#define GROUPS  (CAP / 16)            // 6250 key groups of 16
#define GPW     32                    // key groups per workgroup chunk
#define NCHUNK  ((GROUPS + GPW - 1) / GPW)   // 196 chunks
#define NEG_INF (-__builtin_inff())

// Swizzled B-operand record: per group, 8 K-steps x 2 planes x 32 lanes x 16 bf16.
#define GSTRIDE 8192                  // bf16 elements per group (16 KB)

typedef __attribute__((ext_vector_type(16))) __bf16 v16bf;
typedef __attribute__((ext_vector_type(8)))  __bf16 v8bf;
typedef __attribute__((ext_vector_type(8)))  float  v8f;

// ---- kernel 1a (fast): convert+swizzle keys into WMMA B lane layout, and
// compute ||k||^2, all in ONE fp32 read of the key bank.
// Block = 256 threads = (step s = tid>>5) x (lane = tid&31), one group/block.
__global__ void prep_swz_kernel(const float* __restrict__ keys,
                                __bf16* __restrict__ kswz,
                                float* __restrict__ knorm) {
  __shared__ float red[256];
  const int g    = blockIdx.x;
  const int tid  = threadIdx.x;
  const int s    = tid >> 5;
  const int lane = tid & 31;
  const int r    = lane & 15;
  const int hh   = lane >> 4;

  const float* kr = keys + (size_t)(g * 16 + r) * HID;
  const int k0 = s * 32 + 8 * hh;     // ISA 16-bit operand segment for this lane
  const int k1 = k0 + 16;
  v8f g0 = *(const v8f*)(kr + k0);
  v8f g1 = *(const v8f*)(kr + k1);

  v16bf bh, bl;
  float ss = 0.f;
  #pragma unroll
  for (int e = 0; e < 8; ++e) {
    __bf16 h0 = (__bf16)g0[e];
    __bf16 h1 = (__bf16)g1[e];
    bh[e] = h0;     bl[e]     = (__bf16)(g0[e] - (float)h0);
    bh[8 + e] = h1; bl[8 + e] = (__bf16)(g1[e] - (float)h1);
    ss += g0[e] * g0[e] + g1[e] * g1[e];
  }
  // Coalesced 32B stores: record = [group][step][plane][lane][16 elems]
  __bf16* base = kswz + (size_t)g * GSTRIDE + (size_t)(s * 2) * 512 + lane * 16;
  *(v16bf*)(base)       = bh;
  *(v16bf*)(base + 512) = bl;

  // ||k||^2: threads {s=0..7} x {lane r, r+16} cover row r's 256 elems exactly.
  red[tid] = ss;
  __syncthreads();
  if (tid < 16) {
    float acc = 0.f;
    #pragma unroll
    for (int q = 0; q < 8; ++q)
      acc += red[q * 32 + tid] + red[q * 32 + tid + 16];
    knorm[g * 16 + tid] = acc;
  }
}

// ---------------- kernel 1b (fallback path): key norms only ----------------
__global__ void knorm_kernel(const float* __restrict__ keys,
                             float* __restrict__ knorm) {
  __shared__ float red[256];
  const int row = blockIdx.x;
  const int t   = threadIdx.x;
  float x = keys[(size_t)row * HID + t];
  red[t] = x * x;
  __syncthreads();
  for (int s = 128; s > 0; s >>= 1) {
    if (t < s) red[t] += red[t + s];
    __syncthreads();
  }
  if (t == 0) knorm[row] = red[0];
}

// ---------------- kernel 2: trunk (GEMM + layernorm + tanh -> bf16 hi/lo) --
__global__ void trunk_kernel(const float* __restrict__ obs,
                             const float* __restrict__ act,
                             const float* __restrict__ W,
                             const float* __restrict__ b,
                             const float* __restrict__ gamma,
                             const float* __restrict__ beta,
                             __bf16* __restrict__ phih,
                             __bf16* __restrict__ phil) {
  __shared__ float sx[IN_DIM];
  __shared__ float r1[256];
  __shared__ float r2[256];
  const int row = blockIdx.x;
  const int t   = threadIdx.x;
  if (t < IN_DIM)
    sx[t] = (t < OBS_DIM) ? obs[row * OBS_DIM + t]
                          : act[row * ACT_DIM + (t - OBS_DIM)];
  __syncthreads();
  float acc = b[t];
  #pragma unroll 4
  for (int j = 0; j < IN_DIM; ++j)
    acc = fmaf(sx[j], W[j * HID + t], acc);
  r1[t] = acc;
  r2[t] = acc * acc;
  __syncthreads();
  for (int s = 128; s > 0; s >>= 1) {
    if (t < s) { r1[t] += r1[t + s]; r2[t] += r2[t + s]; }
    __syncthreads();
  }
  const float mu  = r1[0] * (1.0f / HID);
  const float var = r2[0] * (1.0f / HID) - mu * mu;
  const float inv = rsqrtf(var + 1e-5f);
  float p = tanhf((acc - mu) * inv * gamma[t] + beta[t]);
  __bf16 hi = (__bf16)p;
  __bf16 lo = (__bf16)(p - (float)hi);
  phih[(size_t)row * HID + t] = hi;
  phil[(size_t)row * HID + t] = lo;
}

// Shared pieces for the scores kernels ---------------------------------------
__device__ __forceinline__ void load_a_tiles(const __bf16* __restrict__ phih,
                                             const __bf16* __restrict__ phil,
                                             int mtile, int r, int hh,
                                             v16bf (&ahi)[8], v16bf (&alo)[8]) {
  const __bf16* ph = phih + (size_t)(mtile * 16 + r) * HID;
  const __bf16* pl = phil + (size_t)(mtile * 16 + r) * HID;
  #pragma unroll
  for (int s = 0; s < 8; ++s) {
    const int k0 = s * 32 + 8 * hh;
    const int k1 = k0 + 16;
    v8bf a0 = *(const v8bf*)(ph + k0);
    v8bf a1 = *(const v8bf*)(ph + k1);
    v8bf c0 = *(const v8bf*)(pl + k0);
    v8bf c1 = *(const v8bf*)(pl + k1);
    #pragma unroll
    for (int e = 0; e < 8; ++e) {
      ahi[s][e] = a0[e]; ahi[s][8 + e] = a1[e];
      alo[s][e] = c0[e]; alo[s][8 + e] = c1[e];
    }
  }
}

__device__ __forceinline__ void softmax_update(v8f acc, float kn, float vv,
                                               float (&mx)[8], float (&sm)[8],
                                               float (&sv)[8]) {
  #pragma unroll
  for (int i = 0; i < 8; ++i) {
    const float sc = 2.f * acc[i] - kn;            // TAU = 1; phinorm cancels
    const float M  = fmaxf(mx[i], sc);
    const float e0 = __expf(mx[i] - M);            // exp(-inf)=0 on first hit
    const float e1 = __expf(sc - M);
    sm[i] = sm[i] * e0 + e1;
    sv[i] = sv[i] * e0 + e1 * vv;
    mx[i] = M;
  }
}

__device__ __forceinline__ void merge_and_store(int wave, int lane,
                                                float (&mx)[8], float (&sm)[8],
                                                float (&sv)[8],
                                                int mtile, int chunk,
                                                float* __restrict__ pmax,
                                                float* __restrict__ psum,
                                                float* __restrict__ psv) {
  __shared__ float Lmx[4][8][32];
  __shared__ float Lsm[4][8][32];
  __shared__ float Lsv[4][8][32];
  #pragma unroll
  for (int i = 0; i < 8; ++i) {
    Lmx[wave][i][lane] = mx[i];
    Lsm[wave][i][lane] = sm[i];
    Lsv[wave][i][lane] = sv[i];
  }
  __syncthreads();
  if (threadIdx.x < 16) {
    const int m    = threadIdx.x;        // row within tile
    const int half = m >> 3;
    const int slot = m & 7;
    float M = NEG_INF, S = 0.f, V = 0.f;
    for (int w = 0; w < 4; ++w)
      for (int rr = 0; rr < 16; ++rr) {
        const int ln = half * 16 + rr;
        const float m2 = Lmx[w][slot][ln];
        if (m2 == NEG_INF) continue;
        const float Mn = fmaxf(M, m2);
        const float e1 = __expf(M - Mn);
        const float e2 = __expf(m2 - Mn);
        S = S * e1 + Lsm[w][slot][ln] * e2;
        V = V * e1 + Lsv[w][slot][ln] * e2;
        M = Mn;
      }
    const size_t pi = ((size_t)chunk * MT + mtile) * 16 + m;
    pmax[pi] = M; psum[pi] = S; psv[pi] = V;
  }
}

// ---------------- kernel 3a: FAST scores — swizzled bf16 key records -------
// Hot loop: 4 fully-coalesced global_load_b128 + 3 WMMAs per K-step.
__global__ void __launch_bounds__(128)
scores_fast_kernel(const __bf16* __restrict__ kswz,
                   const float* __restrict__ vals,
                   const float* __restrict__ knorm,
                   const __bf16* __restrict__ phih,
                   const __bf16* __restrict__ phil,
                   float* __restrict__ pmax,
                   float* __restrict__ psum,
                   float* __restrict__ psv) {
  const int mtile = blockIdx.x;
  const int chunk = blockIdx.y;
  const int wave  = threadIdx.x >> 5;
  const int lane  = threadIdx.x & 31;
  const int r     = lane & 15;
  const int hh    = lane >> 4;

  v16bf ahi[8], alo[8];
  load_a_tiles(phih, phil, mtile, r, hh, ahi, alo);

  float mx[8], sm[8], sv[8];
  #pragma unroll
  for (int i = 0; i < 8; ++i) { mx[i] = NEG_INF; sm[i] = 0.f; sv[i] = 0.f; }

  #pragma unroll 1
  for (int gi = 0; gi < GPW / 4; ++gi) {
    const int group = chunk * GPW + wave + 4 * gi;   // wave-uniform
    if (group >= GROUPS) break;                      // uniform exit, EXEC full
    const __bf16* gb = kswz + (size_t)group * GSTRIDE + lane * 16;
    if (group + 4 < GROUPS)                          // 32 lanes cover all 16KB
      __builtin_prefetch(kswz + (size_t)(group + 4) * GSTRIDE + lane * 256, 0, 1);

    v8f acc = {};
    #pragma unroll
    for (int s = 0; s < 8; ++s) {
      v16bf bh = *(const v16bf*)(gb + (size_t)(s * 2) * 512);
      v16bf bl = *(const v16bf*)(gb + (size_t)(s * 2 + 1) * 512);
      acc = __builtin_amdgcn_wmma_f32_16x16x32_bf16(false, ahi[s], false, bh,
                                                    (short)0, acc, false, false);
      acc = __builtin_amdgcn_wmma_f32_16x16x32_bf16(false, ahi[s], false, bl,
                                                    (short)0, acc, false, false);
      acc = __builtin_amdgcn_wmma_f32_16x16x32_bf16(false, alo[s], false, bh,
                                                    (short)0, acc, false, false);
    }
    const int key = group * 16 + r;
    softmax_update(acc, knorm[key], vals[key], mx, sm, sv);
  }
  merge_and_store(wave, lane, mx, sm, sv, mtile, chunk, pmax, psum, psv);
}

// ---------------- kernel 3b: FALLBACK scores — convert fp32 keys in-flight -
__global__ void __launch_bounds__(128)
scores_conv_kernel(const float* __restrict__ keys,
                   const float* __restrict__ vals,
                   const float* __restrict__ knorm,
                   const __bf16* __restrict__ phih,
                   const __bf16* __restrict__ phil,
                   float* __restrict__ pmax,
                   float* __restrict__ psum,
                   float* __restrict__ psv) {
  const int mtile = blockIdx.x;
  const int chunk = blockIdx.y;
  const int wave  = threadIdx.x >> 5;
  const int lane  = threadIdx.x & 31;
  const int r     = lane & 15;
  const int hh    = lane >> 4;

  v16bf ahi[8], alo[8];
  load_a_tiles(phih, phil, mtile, r, hh, ahi, alo);

  float mx[8], sm[8], sv[8];
  #pragma unroll
  for (int i = 0; i < 8; ++i) { mx[i] = NEG_INF; sm[i] = 0.f; sv[i] = 0.f; }

  #pragma unroll 1
  for (int gi = 0; gi < GPW / 4; ++gi) {
    const int group = chunk * GPW + wave + 4 * gi;
    if (group >= GROUPS) break;
    const int key = group * 16 + r;
    const float* kr = keys + (size_t)key * HID;
    if (group + 4 < GROUPS)
      __builtin_prefetch(keys + (size_t)(key + 64) * HID, 0, 1);

    v8f acc = {};
    #pragma unroll
    for (int s = 0; s < 8; ++s) {
      const int k0 = s * 32 + 8 * hh;
      const int k1 = k0 + 16;
      v8f g0 = *(const v8f*)(kr + k0);
      v8f g1 = *(const v8f*)(kr + k1);
      v16bf bh, bl;
      #pragma unroll
      for (int e = 0; e < 8; ++e) {
        __bf16 h0 = (__bf16)g0[e];
        __bf16 h1 = (__bf16)g1[e];
        bh[e] = h0;     bl[e]     = (__bf16)(g0[e] - (float)h0);
        bh[8 + e] = h1; bl[8 + e] = (__bf16)(g1[e] - (float)h1);
      }
      acc = __builtin_amdgcn_wmma_f32_16x16x32_bf16(false, ahi[s], false, bh,
                                                    (short)0, acc, false, false);
      acc = __builtin_amdgcn_wmma_f32_16x16x32_bf16(false, ahi[s], false, bl,
                                                    (short)0, acc, false, false);
      acc = __builtin_amdgcn_wmma_f32_16x16x32_bf16(false, alo[s], false, bh,
                                                    (short)0, acc, false, false);
    }
    softmax_update(acc, knorm[key], vals[key], mx, sm, sv);
  }
  merge_and_store(wave, lane, mx, sm, sv, mtile, chunk, pmax, psum, psv);
}

// ---------------- kernel 4: merge chunk partials, write q1 & q2 ------------
__global__ void reduce_kernel(const float* __restrict__ pmax,
                              const float* __restrict__ psum,
                              const float* __restrict__ psv,
                              float* __restrict__ out) {
  const int row = blockIdx.x * blockDim.x + threadIdx.x;
  if (row >= BATCH) return;
  const int mtile = row >> 4, m = row & 15;
  float M = NEG_INF, S = 0.f, V = 0.f;
  for (int c = 0; c < NCHUNK; ++c) {
    const size_t pi = ((size_t)c * MT + mtile) * 16 + m;
    const float m2 = pmax[pi];
    if (m2 == NEG_INF) continue;
    const float Mn = fmaxf(M, m2);
    const float e1 = __expf(M - Mn);
    const float e2 = __expf(m2 - Mn);
    S = S * e1 + psum[pi] * e2;
    V = V * e1 + psv[pi] * e2;
    M = Mn;
  }
  const float q = V / S;
  out[row]         = q;   // q1
  out[BATCH + row] = q;   // q2 (reference computes the identical head twice)
}

// ---------------- launcher -------------------------------------------------
static inline size_t align256(size_t x) { return (x + 255) & ~(size_t)255; }

extern "C" void kernel_launch(void* const* d_in, const int* in_sizes, int n_in,
                              void* d_out, int out_size, void* d_ws, size_t ws_size,
                              hipStream_t stream) {
  const float* obs   = (const float*)d_in[0];
  const float* act   = (const float*)d_in[1];
  const float* W     = (const float*)d_in[2];
  const float* b     = (const float*)d_in[3];
  const float* gamma = (const float*)d_in[4];
  const float* beta  = (const float*)d_in[5];
  const float* keys  = (const float*)d_in[6];
  const float* vals  = (const float*)d_in[7];
  float* out = (float*)d_out;
  (void)in_sizes; (void)n_in; (void)out_size;

  char* ws = (char*)d_ws;
  size_t off = 0;
  float*  knorm = (float*)(ws + off);  off = align256(off + (size_t)CAP * 4);
  __bf16* phih  = (__bf16*)(ws + off); off = align256(off + (size_t)BATCH * HID * 2);
  __bf16* phil  = (__bf16*)(ws + off); off = align256(off + (size_t)BATCH * HID * 2);
  const size_t pelems = (size_t)NCHUNK * MT * 16;
  float* pmax = (float*)(ws + off); off = align256(off + pelems * 4);
  float* psum = (float*)(ws + off); off = align256(off + pelems * 4);
  float* psv  = (float*)(ws + off); off = align256(off + pelems * 4);
  // Swizzled key records (fast path only): GROUPS x 16KB = 102.4 MB.
  __bf16* kswz = (__bf16*)(ws + off);
  off = align256(off + (size_t)GROUPS * GSTRIDE * 2);
  const bool fast = (ws_size >= off);

  trunk_kernel<<<BATCH, 256, 0, stream>>>(obs, act, W, b, gamma, beta, phih, phil);
  if (fast) {
    prep_swz_kernel<<<GROUPS, 256, 0, stream>>>(keys, kswz, knorm);
    scores_fast_kernel<<<dim3(MT, NCHUNK), 128, 0, stream>>>(kswz, vals, knorm,
                                                             phih, phil, pmax, psum, psv);
  } else {
    knorm_kernel<<<CAP, 256, 0, stream>>>(keys, knorm);
    scores_conv_kernel<<<dim3(MT, NCHUNK), 128, 0, stream>>>(keys, vals, knorm,
                                                             phih, phil, pmax, psum, psv);
  }
  reduce_kernel<<<(BATCH + 127) / 128, 128, 0, stream>>>(pmax, psum, psv, out);
}
